// MultiHeadAttention_12541304504488
// MI455X (gfx1250) — matrix operand order repaired and checked by reading, not compiled, not run on hardware
//
#include <hip/hip_runtime.h>
#include <hip/hip_bf16.h>

// ---------------------------------------------------------------------------
// MHA forward for B=4, N=2048, C=1024, H=16, D=64 on gfx1250 (CDNA5, wave32)
// bf16 WMMA (v_wmma_f32_16x16x32_bf16) + flash attention, fp32 accumulate.
// Global->LDS staging uses GLOBAL_LOAD_ASYNC_TO_LDS_B128 (ASYNCcnt) with
// double-buffered LDS tiles for copy/compute overlap.
// ---------------------------------------------------------------------------

typedef __attribute__((ext_vector_type(16))) __bf16 v16bf;
typedef __attribute__((ext_vector_type(4)))  __bf16 v4bf;
typedef __attribute__((ext_vector_type(8)))  float  v8f;

#define EMBED 1024
#define NHEAD 16
#define HDIM  64
#define BATCH 4
#define SEQ   2048
#define M_TOK (BATCH * SEQ)   // 8192 tokens

#define WMMA_BF16(A, B, C) \
  __builtin_amdgcn_wmma_f32_16x16x32_bf16(false, (A), false, (B), (short)0, (C), false, false)

// ---- CDNA5 async global->LDS copy (ASYNCcnt-tracked, no VGPR staging) ------
static __device__ __forceinline__ void async_ld_b128(void* lds_ptr, const void* gptr) {
  // Low 32 bits of a generic LDS address are the byte offset from LDS_BASE.
  const unsigned lds_off = (unsigned)(size_t)lds_ptr;
  asm volatile("global_load_async_to_lds_b128 %0, %1, off"
               :: "v"(lds_off), "v"(gptr) : "memory");
}

static __device__ __forceinline__ void wait_asynccnt0() {
#if __has_builtin(__builtin_amdgcn_s_wait_asynccnt)
  __builtin_amdgcn_s_wait_asynccnt(0);
#else
  asm volatile("s_wait_asynccnt 0x0" ::: "memory");
#endif
}

// ---- Fragment loaders (wave32 layouts per CDNA5 ISA 7.12.2) ----------------
// A 16x32 bf16: lane m = lane&15; lane<16 holds K {0..7,16..23}, lane>=16 adds 8.
// Contiguous 8+8 per lane -> 2x ds_load_b128.
static __device__ __forceinline__ v16bf load_a_frag(const __bf16* p, int ld) {
  const int lane = threadIdx.x & 31;
  const int m    = lane & 15;
  const int koff = (lane >> 4) << 3;
  v16bf a;
#pragma unroll
  for (int i = 0; i < 16; ++i) {
    const int k = ((i < 8) ? i : (i + 8)) + koff;
    a[i] = p[(size_t)m * ld + k];
  }
  return a;
}

// B 32x16 bf16 from a TRANSPOSED source S[n][k] (B[k][n] = S[n][k]).
// Each lane reads K {0..7,16..23} (+8 for lane>=16) from row n -> 2x ds_load_b128.
static __device__ __forceinline__ v16bf load_bT_frag(const __bf16* p, int ld) {
  const int lane = threadIdx.x & 31;
  const int n    = lane & 15;
  const int koff = (lane >> 4) << 3;
  v16bf b;
#pragma unroll
  for (int i = 0; i < 16; ++i) b[i] = p[(size_t)n * ld + koff + ((i < 8) ? i : (i + 8))];
  return b;
}

// ---- fp32 -> bf16 bulk convert (for x, row-major kept) ---------------------
__global__ __launch_bounds__(256) void cvt_f32_bf16(const float4* __restrict__ in,
                                                    __bf16* __restrict__ out, int n4) {
  const int i = blockIdx.x * blockDim.x + threadIdx.x;
  if (i < n4) {
    const float4 v = in[i];
    v4bf o;
    o[0] = (__bf16)v.x; o[1] = (__bf16)v.y; o[2] = (__bf16)v.z; o[3] = (__bf16)v.w;
    *(v4bf*)(out + 4 * (size_t)i) = o;
  }
}

// ---- fp32 -> bf16 convert + transpose: out[n][k] = (bf16) in[k][n] ---------
__global__ __launch_bounds__(256)
void cvt_transpose_bf16(const float* __restrict__ in, __bf16* __restrict__ out,
                        int K, int Nn) {
  __shared__ float t[32][33];
  const int n0 = blockIdx.x * 32, k0 = blockIdx.y * 32;
  const int tx = threadIdx.x, ty = threadIdx.y;   // block (32, 8)
#pragma unroll
  for (int i = ty; i < 32; i += 8)
    t[i][tx] = in[(size_t)(k0 + i) * Nn + n0 + tx];
  __syncthreads();
#pragma unroll
  for (int i = ty; i < 32; i += 8)
    out[(size_t)(n0 + i) * K + k0 + tx] = (__bf16)t[tx][i];
}

// ---- Tiled GEMM: C[M,Nn] = A[M,1024] * W[1024,Nn] + bias -------------------
// W supplied TRANSPOSED: Wt[Nn][1024]. Block tile 128x128, 256 threads (8
// waves); wave w owns rows 16w x all 128 cols (8 WMMAs per k-step, 1 A-frag).
// Double-buffered LDS, async global->LDS staging.
// mode 0: bf16 scatter into Q[B,H,N,D], K[B,H,N,D], Vt[B,H,D,N].
// mode 1: fp32 + bias to Co.
__global__ __launch_bounds__(256)
void gemm_bf16_kernel(const __bf16* __restrict__ A, const __bf16* __restrict__ Wt,
                      const float* __restrict__ bias, int Nn, int mode,
                      __bf16* __restrict__ Qp, __bf16* __restrict__ Kp,
                      __bf16* __restrict__ Vtp, float* __restrict__ Co) {
  __shared__ __bf16 As[2][128][40];   // A rows  [m_local][k_local], padded
  __shared__ __bf16 Bs[2][128][40];   // Wt rows [n_local][k_local], padded

  const int tid  = threadIdx.x;
  const int wave = tid >> 5;
  const int lane = tid & 31;
  const int m0   = wave * 16;
  const int blkM = blockIdx.y * 128;
  const int blkN = blockIdx.x * 128;

  const int lr = tid >> 1, lc = (tid & 1) * 16;   // 16 bf16 (2x b128) / thread

  const __bf16* ga = &A [(size_t)(blkM + lr) * EMBED + lc];
  const __bf16* gb = &Wt[(size_t)(blkN + lr) * EMBED + lc];

  // prime buffer 0 with k0 = 0
  async_ld_b128(&As[0][lr][lc],     ga);
  async_ld_b128(&As[0][lr][lc + 8], ga + 8);
  async_ld_b128(&Bs[0][lr][lc],     gb);
  async_ld_b128(&Bs[0][lr][lc + 8], gb + 8);

  v8f acc[8] = {};

  for (int k0 = 0; k0 < EMBED; k0 += 32) {
    const int cur = (k0 >> 5) & 1;
    wait_asynccnt0();          // this wave's async writes into buf `cur` landed
    __syncthreads();           // everyone's landed; prev reads of buf cur^1 done
    if (k0 + 32 < EMBED) {     // issue next tile into the other buffer
      const int kn = k0 + 32;
      async_ld_b128(&As[cur ^ 1][lr][lc],     ga + kn);
      async_ld_b128(&As[cur ^ 1][lr][lc + 8], ga + kn + 8);
      async_ld_b128(&Bs[cur ^ 1][lr][lc],     gb + kn);
      async_ld_b128(&Bs[cur ^ 1][lr][lc + 8], gb + kn + 8);
    }

    const v16bf a = load_a_frag(&As[cur][m0][0], 40);
#pragma unroll
    for (int t = 0; t < 8; ++t)
      acc[t] = WMMA_BF16(a, load_bT_frag(&Bs[cur][16 * t][0], 40), acc[t]);
  }

  // Epilogue: C layout VGPR r -> row r + 8*(lane>=16), col = lane&15.
#pragma unroll
  for (int t = 0; t < 8; ++t) {
#pragma unroll
    for (int r = 0; r < 8; ++r) {
      const int grow = blkM + m0 + r + ((lane >> 4) << 3);
      const int gcol = blkN + 16 * t + (lane & 15);
      const float c  = acc[t][r] + bias[gcol];
      if (mode == 0) {
        const int b = grow >> 11, ntok = grow & 2047;     // token -> (batch, seq)
        const int s = gcol >> 10, hc = gcol & 1023;       // col -> (q/k/v, head, d)
        const int h = hc >> 6, d = hc & 63;
        const size_t bh = (size_t)b * NHEAD + h;
        if (s == 0)      Qp [(bh * SEQ + ntok) * HDIM + d]  = (__bf16)c;
        else if (s == 1) Kp [(bh * SEQ + ntok) * HDIM + d]  = (__bf16)c;
        else             Vtp[(bh * HDIM + d) * SEQ + ntok]  = (__bf16)c;  // transposed V
      } else {
        Co[(size_t)grow * Nn + gcol] = c;
      }
    }
  }
}

// ---- Flash attention --------------------------------------------------------
// Block: 128 threads = 4 waves, each wave owns a 16-row query tile (64 rows/block).
// K chunk staged as [32 keys][64 d]; V chunk staged transposed [64 d][32 keys].
// Double-buffered chunks, async global->LDS staging.
__global__ __launch_bounds__(128)
void flash_attn_kernel(const __bf16* __restrict__ Qp, const __bf16* __restrict__ Kp,
                       const __bf16* __restrict__ Vtp, __bf16* __restrict__ Op) {
  __shared__ __bf16 Ks [2][32][72];   // K rows (j local, d), padded
  __shared__ __bf16 VsT[2][64][40];   // V transposed (d, j local), padded
  __shared__ __bf16 Ps [4][16][40];   // per-wave softmax tile 16x32, padded

  const int tid  = threadIdx.x;
  const int wave = tid >> 5;
  const int lane = tid & 31;
  const int h = blockIdx.y, b = blockIdx.z;
  const int row0 = blockIdx.x * 64 + wave * 16;
  const size_t bh    = (size_t)b * NHEAD + h;
  const size_t kbase = bh * SEQ * HDIM;   // K: [N][D]
  const size_t vbase = bh * HDIM * SEQ;   // Vt: [D][N]
  const float  scale = 0.125f;            // 1/sqrt(64)

  const v16bf q0 = load_a_frag(Qp + kbase + (size_t)row0 * HDIM, HDIM);       // d 0..31
  const v16bf q1 = load_a_frag(Qp + kbase + (size_t)row0 * HDIM + 32, HDIM);  // d 32..63

  v8f o[4] = {};
  v8f m_run, l_run;
#pragma unroll
  for (int r = 0; r < 8; ++r) { m_run[r] = -1e30f; l_run[r] = 0.0f; }

  const int kr = tid >> 2, kc = (tid & 3) * 16;  // K chunk map: 32 rows x 64 d
  const int vr = tid >> 1, vc = (tid & 1) * 16;  // V chunk map: 64 rows x 32 j

  const __bf16* gk = &Kp [kbase + (size_t)kr * HDIM + kc];  // advance by 32*HDIM / iter
  const __bf16* gv = &Vtp[vbase + (size_t)vr * SEQ + vc];   // advance by 32 / iter

  // prime buffer 0 with j0 = 0
  async_ld_b128(&Ks[0][kr][kc],      gk);
  async_ld_b128(&Ks[0][kr][kc + 8],  gk + 8);
  async_ld_b128(&VsT[0][vr][vc],     gv);
  async_ld_b128(&VsT[0][vr][vc + 8], gv + 8);

  for (int j0 = 0; j0 < SEQ; j0 += 32) {
    const int cur = (j0 >> 5) & 1;
    wait_asynccnt0();
    __syncthreads();
    if (j0 + 32 < SEQ) {
      const __bf16* gkn = gk + (size_t)(j0 + 32) * HDIM;
      const __bf16* gvn = gv + (j0 + 32);
      async_ld_b128(&Ks[cur ^ 1][kr][kc],      gkn);
      async_ld_b128(&Ks[cur ^ 1][kr][kc + 8],  gkn + 8);
      async_ld_b128(&VsT[cur ^ 1][vr][vc],     gvn);
      async_ld_b128(&VsT[cur ^ 1][vr][vc + 8], gvn + 8);
    }

    // S tiles: S[i][j] = sum_d Q[i][d] * K[j][d]   (B-frag = transposed K rows)
    v8f s0 = {}, s1 = {};
    s0 = WMMA_BF16(q0, load_bT_frag(&Ks[cur][0][0],   72), s0);
    s0 = WMMA_BF16(q1, load_bT_frag(&Ks[cur][0][32],  72), s0);
    s1 = WMMA_BF16(q0, load_bT_frag(&Ks[cur][16][0],  72), s1);
    s1 = WMMA_BF16(q1, load_bT_frag(&Ks[cur][16][32], 72), s1);

    // scale + running row max (16 cols of a row live in one 16-lane group)
    v8f curmax;
#pragma unroll
    for (int r = 0; r < 8; ++r) {
      s0[r] *= scale; s1[r] *= scale;
      curmax[r] = fmaxf(s0[r], s1[r]);
    }
#pragma unroll
    for (int mk = 1; mk <= 8; mk <<= 1)
#pragma unroll
      for (int r = 0; r < 8; ++r)
        curmax[r] = fmaxf(curmax[r], __shfl_xor(curmax[r], mk, 32));

    v8f psum;
#pragma unroll
    for (int r = 0; r < 8; ++r) {
      const float mn    = fmaxf(m_run[r], curmax[r]);
      const float alpha = __expf(m_run[r] - mn);
      m_run[r] = mn;
      l_run[r] *= alpha;
      o[0][r] *= alpha; o[1][r] *= alpha; o[2][r] *= alpha; o[3][r] *= alpha;
      s0[r]  = __expf(s0[r] - mn);
      s1[r]  = __expf(s1[r] - mn);
      psum[r] = s0[r] + s1[r];
    }
#pragma unroll
    for (int mk = 1; mk <= 8; mk <<= 1)
#pragma unroll
      for (int r = 0; r < 8; ++r)
        psum[r] += __shfl_xor(psum[r], mk, 32);
#pragma unroll
    for (int r = 0; r < 8; ++r) l_run[r] += psum[r];

    // stage P (bf16) into LDS to re-enter A-fragment layout
    const int prow_hi = (lane >> 4) << 3;
    const int pcol    = lane & 15;
#pragma unroll
    for (int r = 0; r < 8; ++r) {
      Ps[wave][r + prow_hi][pcol]      = (__bf16)s0[r];
      Ps[wave][r + prow_hi][pcol + 16] = (__bf16)s1[r];
    }
    __syncthreads();

    // O += P * V  (4 d-tiles of 16; B-frag from transposed V rows, contiguous)
    const v16bf pa = load_a_frag(&Ps[wave][0][0], 40);
#pragma unroll
    for (int t = 0; t < 4; ++t)
      o[t] = WMMA_BF16(pa, load_bT_frag(&VsT[cur][16 * t][0], 40), o[t]);
  }

  // normalize and write O as [B*N, C] bf16 (row-major for the out projection)
#pragma unroll
  for (int t = 0; t < 4; ++t) {
#pragma unroll
    for (int r = 0; r < 8; ++r) {
      const int grow = row0 + r + ((lane >> 4) << 3);
      const int gcol = h * HDIM + t * 16 + (lane & 15);
      const float val = o[t][r] / l_run[r];
      Op[((size_t)b * SEQ + grow) * EMBED + gcol] = (__bf16)val;
    }
  }
}

// ---------------------------------------------------------------------------
extern "C" void kernel_launch(void* const* d_in, const int* in_sizes, int n_in,
                              void* d_out, int out_size, void* d_ws, size_t ws_size,
                              hipStream_t stream) {
  (void)in_sizes; (void)n_in; (void)out_size; (void)ws_size;
  const float* x     = (const float*)d_in[0];
  const float* w_qkv = (const float*)d_in[1];
  const float* b_qkv = (const float*)d_in[2];
  const float* w_out = (const float*)d_in[3];
  const float* b_out = (const float*)d_in[4];
  float* out = (float*)d_out;

  char* ws = (char*)d_ws;
  size_t off = 0;
  __bf16* x_bf   = (__bf16*)(ws + off); off += (size_t)M_TOK * EMBED * 2;      // 16 MB
  __bf16* wqkvT  = (__bf16*)(ws + off); off += (size_t)EMBED * 3 * EMBED * 2;  //  6 MB (transposed)
  __bf16* woutT  = (__bf16*)(ws + off); off += (size_t)EMBED * EMBED * 2;      //  2 MB (transposed)
  __bf16* Qh     = (__bf16*)(ws + off); off += (size_t)M_TOK * EMBED * 2;      // 16 MB [B,H,N,D]
  __bf16* Kh     = (__bf16*)(ws + off); off += (size_t)M_TOK * EMBED * 2;      // 16 MB [B,H,N,D]
  __bf16* Vt     = (__bf16*)(ws + off); off += (size_t)M_TOK * EMBED * 2;      // 16 MB [B,H,D,N]
  __bf16* Ob     = (__bf16*)(ws + off); off += (size_t)M_TOK * EMBED * 2;      // 16 MB [B*N, C]

  // 1) converts: x row-major; weights converted + transposed
  {
    const int n4 = M_TOK * EMBED / 4;
    cvt_f32_bf16<<<(n4 + 255) / 256, 256, 0, stream>>>((const float4*)x, x_bf, n4);
  }
  cvt_transpose_bf16<<<dim3(3 * EMBED / 32, EMBED / 32), dim3(32, 8), 0, stream>>>(
      w_qkv, wqkvT, EMBED, 3 * EMBED);
  cvt_transpose_bf16<<<dim3(EMBED / 32, EMBED / 32), dim3(32, 8), 0, stream>>>(
      w_out, woutT, EMBED, EMBED);

  // 2) QKV projection: [8192,1024] x [1024,3072] -> Q, K, Vt (per head)
  gemm_bf16_kernel<<<dim3(3 * EMBED / 128, M_TOK / 128), 256, 0, stream>>>(
      x_bf, wqkvT, b_qkv, 3 * EMBED, 0, Qh, Kh, Vt, nullptr);

  // 3) Flash attention -> Ob [B*N, C] bf16
  flash_attn_kernel<<<dim3(SEQ / 64, NHEAD, BATCH), 128, 0, stream>>>(Qh, Kh, Vt, Ob);

  // 4) Output projection: [8192,1024] x [1024,1024] + b_out -> d_out fp32
  gemm_bf16_kernel<<<dim3(EMBED / 128, M_TOK / 128), 256, 0, stream>>>(
      Ob, woutT, b_out, EMBED, 1, nullptr, nullptr, nullptr, out);
}
